// _ProxyAttentionBlock_70257075028693
// MI455X (gfx1250) — matrix operand order repaired and checked by reading, not compiled, not run on hardware
//
#include <hip/hip_runtime.h>
#include <hip/hip_bf16.h>

// ---------------------------------------------------------------------------
// ProxyAttentionBlock for MI455X (gfx1250), wave32 + WMMA bf16.
// B=8, CIN=512, CK=256, P=128, H=W=96 -> N=9216.
// All GEMMs run on v_wmma_f32_16x16x32_bf16 (fp32 sources converted to bf16
// while staging through LDS; fp32 accumulation). LDS is double-buffered; bf16
// k-contiguous tiles are staged with gfx1250 global_load_async_to_lds_b128
// (ASYNCcnt) when the toolchain exposes the builtin. Split-K + f32 atomics
// for the small-output big-K GEMM (proxy_ctx).
// ---------------------------------------------------------------------------

typedef __attribute__((ext_vector_type(16))) __bf16 v16bf;
typedef __attribute__((ext_vector_type(8)))  float  v8f;

#define BN_SCALE_C 0.9999950000374997f   // 1/sqrt(1+1e-5)
#define EPS_NRM_C  1e-12f

enum { DT_F32 = 0, DT_BF16 = 1 };
enum { LAY_NN = 0, LAY_NT = 1 };           // NN: B is KxN (n contig). NT: B is NxK (k contig).
enum { EPI_NONE = 0, EPI_BNRELU = 1, EPI_RELU = 2, EPI_BIAS = 3 };

#define LDST 40   // LDS row stride in bf16 elements for 32-wide K tiles (+8 pad)

// ---- gfx1250 async global->LDS copy (ASYNCcnt), guarded by __has_builtin ----
#if defined(__has_builtin)
#if __has_builtin(__builtin_amdgcn_global_load_async_to_lds_b128)
#define HAS_ASYNC_LDS 1
#endif
#endif
#ifndef HAS_ASYNC_LDS
#define HAS_ASYNC_LDS 0
#endif

#if HAS_ASYNC_LDS
typedef int v4i_ __attribute__((vector_size(4 * sizeof(int))));
__device__ inline void async_cp16(const void* g, __bf16* l)
{
  // param0: v4i in device (AS1) space; param1: v4i in LDS (AS3) space.
  // Cast through uintptr: AS1 pointers share the generic bit pattern; the low
  // 32 bits of a generic LDS pointer are the LDS byte offset (ISA 10.2).
  __builtin_amdgcn_global_load_async_to_lds_b128(
      (__attribute__((address_space(1))) v4i_*)(size_t)g,
      (__attribute__((address_space(3))) v4i_*)(size_t)l,
      0, 0);
}
#endif
__device__ inline void wait_async_zero()
{
#if HAS_ASYNC_LDS
  asm volatile("s_wait_asynccnt 0x0" ::: "memory");
#endif
}

// ---- tile loaders: stage a 128x32 tile into LDS as bf16, [row][k] layout ----

template <int DT>
__device__ inline void load_kcontig(__bf16* lds, const void* src, int ld, int tid)
{
#pragma unroll
  for (int it = 0; it < 2; ++it) {
    int linear = it * 256 + tid;          // 0..511
    int r  = linear >> 2;                 // 0..127
    int kg = (linear & 3) << 3;           // 0,8,16,24
    __bf16* dst = lds + r * LDST + kg;
    if (DT == DT_F32) {
      const float* p = (const float*)src + (size_t)r * ld + kg;
      float4 f0 = *(const float4*)(p);
      float4 f1 = *(const float4*)(p + 4);
      union { uint4 u; __bf16 h[8]; } pk;
      pk.h[0] = (__bf16)f0.x; pk.h[1] = (__bf16)f0.y;
      pk.h[2] = (__bf16)f0.z; pk.h[3] = (__bf16)f0.w;
      pk.h[4] = (__bf16)f1.x; pk.h[5] = (__bf16)f1.y;
      pk.h[6] = (__bf16)f1.z; pk.h[7] = (__bf16)f1.w;
      *(uint4*)dst = pk.u;
    } else {
      const __bf16* p = (const __bf16*)src + (size_t)r * ld + kg;
#if HAS_ASYNC_LDS
      async_cp16(p, dst);                 // global_load_async_to_lds_b128
#else
      *(uint4*)dst = *(const uint4*)p;
#endif
    }
  }
}

// source is 32(K) x 128(N), n contiguous; store transposed -> lds[n][k]
template <int DT>
__device__ inline void load_ncontig(__bf16* lds, const void* src, int ld, int tid)
{
#pragma unroll
  for (int it = 0; it < 2; ++it) {
    int linear = it * 256 + tid;          // 0..511
    int k  = linear >> 4;                 // 0..31
    int ng = (linear & 15) << 3;          // 0..120
    if (DT == DT_F32) {
      const float* p = (const float*)src + (size_t)k * ld + ng;
      float4 f0 = *(const float4*)(p);
      float4 f1 = *(const float4*)(p + 4);
      lds[(ng + 0) * LDST + k] = (__bf16)f0.x;
      lds[(ng + 1) * LDST + k] = (__bf16)f0.y;
      lds[(ng + 2) * LDST + k] = (__bf16)f0.z;
      lds[(ng + 3) * LDST + k] = (__bf16)f0.w;
      lds[(ng + 4) * LDST + k] = (__bf16)f1.x;
      lds[(ng + 5) * LDST + k] = (__bf16)f1.y;
      lds[(ng + 6) * LDST + k] = (__bf16)f1.z;
      lds[(ng + 7) * LDST + k] = (__bf16)f1.w;
    } else {
      const __bf16* p = (const __bf16*)src + (size_t)k * ld + ng;
      union { uint4 u; __bf16 h[8]; } pk;
      pk.u = *(const uint4*)p;
#pragma unroll
      for (int j = 0; j < 8; ++j) lds[(ng + j) * LDST + k] = pk.h[j];
    }
  }
}

// ---- per-lane fragment gathers (ISA 7.12.2 layouts, wave32) ----

// A 16x32 bf16: k = (j>>2)*16 + half*8 + (j&3)*2
__device__ inline v16bf frag_a_load(const __bf16* lds, int row, int half)
{
  union { v16bf v; unsigned u[8]; } f;
  const unsigned* base = (const unsigned*)(lds + row * LDST);
#pragma unroll
  for (int j = 0; j < 8; ++j) {
    int k = ((j >> 2) << 4) + (half << 3) + ((j & 3) << 1);
    f.u[j] = base[k >> 1];
  }
  return f.v;
}

// B 32x16 bf16 (k-contig per output column in LDS): k = (j>>2)*16 + half*8 + (j&3)*2
__device__ inline v16bf frag_b_load(const __bf16* lds, int row, int half)
{
  union { v16bf v; unsigned u[8]; } f;
  const unsigned* base = (const unsigned*)(lds + row * LDST);
#pragma unroll
  for (int j = 0; j < 8; ++j) {
    int k = ((j >> 2) << 4) + (half << 3) + ((j & 3) << 1);
    f.u[j] = base[k >> 1];
  }
  return f.v;
}

// ---- the GEMM kernel: C(MxN) += A(MxK) * op(B); 128x128 tile / workgroup ----

template <int ADT, int BDT, int BLAY, int EPI, int ODT, bool SPLITK>
__global__ __launch_bounds__(256)
void gemm_wmma(const void* __restrict__ Ag, const void* __restrict__ Bg,
               void* __restrict__ Cg,
               int M, int N, int K, int lda, int ldb, int ldc,
               long long sAb, long long sBb, long long sCb, int ksplits,
               const float* __restrict__ bias, const float* __restrict__ gma,
               const float* __restrict__ bta)
{
  __shared__ __bf16 ldsA[2][128 * LDST];
  __shared__ __bf16 ldsB[2][128 * LDST];

  const int tid  = threadIdx.x;
  const int wave = tid >> 5;
  const int lane = tid & 31;
  const int half = lane >> 4;
  const int lr   = lane & 15;
  const int wm   = (wave >> 1) * 32;     // wave M offset within 128 tile
  const int wn   = (wave & 1) * 64;      // wave N offset within 128 tile

  const int tileM = blockIdx.x * 128;
  const int tileN = blockIdx.y * 128;
  int bz = blockIdx.z, split = 0;
  if (SPLITK) { split = bz % ksplits; bz /= ksplits; }

  const int kChunk = SPLITK ? (K / ksplits) : K;
  const int k0 = split * kChunk;
  const int iters = kChunk / 32;

  const size_t esA = (ADT == DT_F32) ? 4 : 2;
  const size_t esB = (BDT == DT_F32) ? 4 : 2;
  const char* Abase = (const char*)Ag + (size_t)bz * (size_t)sAb * esA;
  const char* Bbase = (const char*)Bg + (size_t)bz * (size_t)sBb * esB;

  // does this instantiation issue ASYNCcnt ops?
  const bool useAsync =
      HAS_ASYNC_LDS && (ADT == DT_BF16 || (BDT == DT_BF16 && BLAY == LAY_NT));

  auto stage = [&](int kk, __bf16* bA, __bf16* bB) {
    const char* aSrc = Abase + ((size_t)tileM * lda + kk) * esA;
    load_kcontig<ADT>(bA, aSrc, lda, tid);
    if (BLAY == LAY_NT) {
      const char* bSrc = Bbase + ((size_t)tileN * ldb + kk) * esB;
      load_kcontig<BDT>(bB, bSrc, ldb, tid);
    } else {
      const char* bSrc = Bbase + ((size_t)kk * ldb + tileN) * esB;
      load_ncontig<BDT>(bB, bSrc, ldb, tid);
      __builtin_prefetch(bSrc + 32 * (size_t)ldb * esB, 0, 1);  // global_prefetch_b8
    }
  };

  v8f acc[2][4];
#pragma unroll
  for (int mt = 0; mt < 2; ++mt)
#pragma unroll
    for (int nt = 0; nt < 4; ++nt)
#pragma unroll
      for (int r = 0; r < 8; ++r) acc[mt][nt][r] = 0.0f;

  stage(k0, ldsA[0], ldsB[0]);

  for (int i = 0; i < iters; ++i) {
    const int cur = i & 1, nxt = cur ^ 1;
    if (useAsync) wait_async_zero();     // s_wait_asynccnt 0 (this wave's tile)
    __syncthreads();                     // tile i visible; buf[nxt] free to fill
    if (i + 1 < iters) stage(k0 + (i + 1) * 32, ldsA[nxt], ldsB[nxt]);

    v16bf a0 = frag_a_load(ldsA[cur], wm + lr, half);
    v16bf a1 = frag_a_load(ldsA[cur], wm + 16 + lr, half);
#pragma unroll
    for (int nt = 0; nt < 4; ++nt) {
      v16bf bfv = frag_b_load(ldsB[cur], wn + nt * 16 + lr, half);
      acc[0][nt] = __builtin_amdgcn_wmma_f32_16x16x32_bf16(
          false, a0, false, bfv, (short)0, acc[0][nt], false, false);
      acc[1][nt] = __builtin_amdgcn_wmma_f32_16x16x32_bf16(
          false, a1, false, bfv, (short)0, acc[1][nt], false, false);
    }
  }

  float*  Cf = (float*)Cg;
  __bf16* Ch = (__bf16*)Cg;
  const size_t cb = (size_t)bz * (size_t)sCb;
#pragma unroll
  for (int mt = 0; mt < 2; ++mt)
#pragma unroll
    for (int nt = 0; nt < 4; ++nt)
#pragma unroll
      for (int r = 0; r < 8; ++r) {
        int mg = tileM + wm + mt * 16 + half * 8 + r;
        int ng = tileN + wn + nt * 16 + lr;
        float v = acc[mt][nt][r];
        if (EPI == EPI_BNRELU) {
          v = fmaxf((v + bias[mg]) * (BN_SCALE_C * gma[mg]) + bta[mg], 0.0f);
        } else if (EPI == EPI_RELU) {
          v = fmaxf(v, 0.0f);
        } else if (EPI == EPI_BIAS) {
          v = v + bias[mg];
        }
        size_t off = cb + (size_t)mg * ldc + ng;
        if (SPLITK)             atomicAdd(Cf + off, v);
        else if (ODT == DT_F32) Cf[off] = v;
        else                    Ch[off] = (__bf16)v;
      }
}

// ---- column L2 normalize over C rows of a (C x Nn) fp32 map; bf16 out ------

template <bool TRANS>
__global__ __launch_bounds__(256)
void colnorm_l2(const float* __restrict__ X, __bf16* __restrict__ Y,
                int C, int Nn, long long sXb, long long sYb)
{
  int n = blockIdx.x * 256 + threadIdx.x;
  int b = blockIdx.y;
  if (n >= Nn) return;
  const float* x = X + (size_t)b * sXb + n;
  float a = 0.0f;
  for (int c = 0; c < C; ++c) { float v = x[(size_t)c * Nn]; a += v * v; }
  float s = 1.0f / fmaxf(sqrtf(a), EPS_NRM_C);
  __bf16* y = Y + (size_t)b * sYb;
  for (int c = 0; c < C; ++c) {
    float v = x[(size_t)c * Nn] * s;
    if (TRANS) y[(size_t)n * C + c] = (__bf16)v;
    else       y[(size_t)c * Nn + n] = (__bf16)v;
  }
}

// ---- row L1 normalize: rows of length L, fp32 in -> bf16 out ---------------

__global__ __launch_bounds__(256)
void rowl1(const float* __restrict__ X, __bf16* __restrict__ Y, int L,
           long long sXb, long long sYb)
{
  __shared__ float red[256];
  const float* x = X + (size_t)blockIdx.y * sXb + (size_t)blockIdx.x * L;
  float a = 0.0f;
  for (int i = threadIdx.x; i < L; i += 256) a += fabsf(x[i]);
  red[threadIdx.x] = a;
  __syncthreads();
  for (int s = 128; s > 0; s >>= 1) {
    if ((int)threadIdx.x < s) red[threadIdx.x] += red[threadIdx.x + s];
    __syncthreads();
  }
  float sc = 1.0f / fmaxf(red[0], EPS_NRM_C);
  __bf16* y = Y + (size_t)blockIdx.y * sYb + (size_t)blockIdx.x * L;
  for (int i = threadIdx.x; i < L; i += 256) y[i] = (__bf16)(x[i] * sc);
}

// ---- misc: transpose+convert, zero ----------------------------------------

__global__ __launch_bounds__(256)
void transpose_cvt(const float* __restrict__ X, __bf16* __restrict__ Y,
                   int R, int Cc, long long sXb, long long sYb)
{
  int b = blockIdx.y;
  for (int i = blockIdx.x * 256 + threadIdx.x; i < R * Cc; i += gridDim.x * 256) {
    int r = i / Cc, c = i % Cc;
    Y[(size_t)b * sYb + (size_t)c * R + r] = (__bf16)X[(size_t)b * sXb + i];
  }
}

__global__ __launch_bounds__(256)
void zero_f32(float* __restrict__ p, long long count)
{
  for (long long i = (long long)blockIdx.x * 256 + threadIdx.x; i < count;
       i += (long long)gridDim.x * 256)
    p[i] = 0.0f;
}

// ---------------------------------------------------------------------------

extern "C" void kernel_launch(void* const* d_in, const int* in_sizes, int n_in,
                              void* d_out, int out_size, void* d_ws, size_t ws_size,
                              hipStream_t stream)
{
  (void)in_sizes; (void)n_in; (void)out_size; (void)ws_size;

  constexpr int Bb = 8, CIN = 512, CK = 256, Pp = 128, Nn = 96 * 96; // 9216

  const float* x     = (const float*)d_in[0];
  const float* pfeat = (const float*)d_in[1];
  const float* fk_w  = (const float*)d_in[2];
  const float* fk_b  = (const float*)d_in[3];
  const float* fk_g  = (const float*)d_in[4];
  const float* fk_be = (const float*)d_in[5];
  const float* fq_w  = (const float*)d_in[6];
  const float* fq_b  = (const float*)d_in[7];
  const float* fq_g  = (const float*)d_in[8];
  const float* fq_be = (const float*)d_in[9];
  const float* fp1_w = (const float*)d_in[10];
  const float* fp1_b = (const float*)d_in[11];
  const float* fp1_g = (const float*)d_in[12];
  const float* fp1_be= (const float*)d_in[13];
  const float* fp2_w = (const float*)d_in[14];
  const float* fp2_b = (const float*)d_in[15];
  const float* fp2_g = (const float*)d_in[16];
  const float* fp2_be= (const float*)d_in[17];
  const float* W_w   = (const float*)d_in[18];
  const float* W_b   = (const float*)d_in[19];

  // ---- workspace layout (bytes), with reuse ----
  char* ws = (char*)d_ws;
  size_t o = 0;
  auto alloc = [&](size_t bytes) { size_t r = o; o += (bytes + 255) & ~(size_t)255; return r; };
  const size_t R1 = alloc((size_t)Bb * CK * Nn * 4);   // key32 -> q32 -> x_ctx_bf
  const size_t R2 = alloc((size_t)Bb * CK * Nn * 2);   // key_bf -> q_bf
  const size_t R3 = alloc((size_t)Bb * Pp * Nn * 4);   // sim1 raw -> sim2 raw (f32)
  const size_t R4 = alloc((size_t)Bb * Pp * Nn * 2);   // sim1_bf -> sim2_bf
  const size_t PR1 = alloc((size_t)CK * Pp * 4);
  const size_t PR2 = alloc((size_t)CK * Pp * 4);
  const size_t PXT = alloc((size_t)Pp * CK * 2);       // proxyT bf16 (P x CK)
  const size_t PCX = alloc((size_t)Bb * Pp * CK * 4);  // proxy_ctx f32 (B,P,CK)
  const size_t PCT = alloc((size_t)Bb * CK * Pp * 2);  // pcT bf16 (B,CK,P)

  float*  R1f = (float*)(ws + R1);  __bf16* R1h = (__bf16*)(ws + R1);
  __bf16* R2h = (__bf16*)(ws + R2);
  float*  R3f = (float*)(ws + R3);
  __bf16* R4h = (__bf16*)(ws + R4);
  float*  pr1 = (float*)(ws + PR1);
  float*  pr2 = (float*)(ws + PR2);
  __bf16* pxT = (__bf16*)(ws + PXT);
  float*  pcx = (float*)(ws + PCX);
  __bf16* pcT = (__bf16*)(ws + PCT);

  const long long sKey = (long long)CK * Nn;   // 256*9216 elements
  const long long sSim = (long long)Pp * Nn;   // 128*9216
  const long long sX   = (long long)CIN * Nn;  // 512*9216
  const long long sPC  = (long long)Pp * CK;   // 128*256

  // --- proxy head (tiny): pr1 = BNReLU(fp1_w @ proxy_feat) ---
  gemm_wmma<DT_F32, DT_F32, LAY_NN, EPI_BNRELU, DT_F32, false>
      <<<dim3(2, 1, 1), 256, 0, stream>>>(fp1_w, pfeat, pr1,
          CK, Pp, CIN, CIN, Pp, Pp, 0, 0, 0, 1, fp1_b, fp1_g, fp1_be);
  // --- pr2 = BNReLU(fp2_w @ pr1) ---
  gemm_wmma<DT_F32, DT_F32, LAY_NN, EPI_BNRELU, DT_F32, false>
      <<<dim3(2, 1, 1), 256, 0, stream>>>(fp2_w, pr1, pr2,
          CK, Pp, CK, CK, Pp, Pp, 0, 0, 0, 1, fp2_b, fp2_g, fp2_be);
  // --- proxyT = l2norm over CK, transposed (P x CK) bf16 ---
  colnorm_l2<true><<<dim3(1, 1), 256, 0, stream>>>(pr2, pxT, CK, Pp, 0, 0);

  // --- key32 = BNReLU(fk_w @ x)  (per batch 256x9216, K=512) ---
  gemm_wmma<DT_F32, DT_F32, LAY_NN, EPI_BNRELU, DT_F32, false>
      <<<dim3(2, 72, 8), 256, 0, stream>>>(fk_w, x, R1f,
          CK, Nn, CIN, CIN, Nn, Nn, 0, sX, sKey, 1, fk_b, fk_g, fk_be);
  // --- key_bf = l2norm over channels ---
  colnorm_l2<false><<<dim3(36, 8), 256, 0, stream>>>(R1f, R2h, CK, Nn, sKey, sKey);

  // --- sim1 = relu(proxyT @ key_bf)  (128x9216, K=256) ---
  gemm_wmma<DT_BF16, DT_BF16, LAY_NN, EPI_RELU, DT_F32, false>
      <<<dim3(1, 72, 8), 256, 0, stream>>>(pxT, R2h, R3f,
          Pp, Nn, CK, CK, Nn, Nn, 0, sKey, sSim, 1, nullptr, nullptr, nullptr);
  // --- sim1_bf = L1 normalize over N ---
  rowl1<<<dim3(Pp, Bb), 256, 0, stream>>>(R3f, R4h, Nn, sSim, sSim);

  // --- proxy_ctx = sim1_bf @ key_bf^T  (128x256, K=9216) split-K + atomics ---
  zero_f32<<<dim3(256), 256, 0, stream>>>(pcx, (long long)Bb * Pp * CK);
  gemm_wmma<DT_BF16, DT_BF16, LAY_NT, EPI_NONE, DT_F32, true>
      <<<dim3(1, 2, 64), 256, 0, stream>>>(R4h, R2h, pcx,
          Pp, CK, Nn, Nn, Nn, CK, sSim, sKey, sPC, 8, nullptr, nullptr, nullptr);

  // --- q32 = BNReLU(fq_w @ key_bf)  (256x9216, K=256) ---
  gemm_wmma<DT_F32, DT_BF16, LAY_NN, EPI_BNRELU, DT_F32, false>
      <<<dim3(2, 72, 8), 256, 0, stream>>>(fq_w, R2h, R1f,
          CK, Nn, CK, CK, Nn, Nn, 0, sKey, sKey, 1, fq_b, fq_g, fq_be);
  // --- q_bf = l2norm over channels, transposed to (N x CK) ---
  colnorm_l2<true><<<dim3(36, 8), 256, 0, stream>>>(R1f, R2h, CK, Nn, sKey, sKey);

  // --- sim2 = relu(q_bf @ proxyT^T)  (9216x128, K=256) ---
  gemm_wmma<DT_BF16, DT_BF16, LAY_NT, EPI_RELU, DT_F32, false>
      <<<dim3(72, 1, 8), 256, 0, stream>>>(R2h, pxT, R3f,
          Nn, Pp, CK, CK, CK, Pp, sKey, 0, sSim, 1, nullptr, nullptr, nullptr);
  // --- sim2_bf = L1 normalize over P ---
  rowl1<<<dim3(Nn, Bb), 256, 0, stream>>>(R3f, R4h, Pp, sSim, sSim);

  // --- pcT = transpose(proxy_ctx) bf16  (B,CK,P) ---
  transpose_cvt<<<dim3(32, 8), 256, 0, stream>>>(pcx, pcT, Pp, CK, sPC, sPC);

  // --- x_ctx_bf = pcT @ sim2_bf^T  (256x9216, K=128), bf16 out ---
  gemm_wmma<DT_BF16, DT_BF16, LAY_NT, EPI_NONE, DT_BF16, false>
      <<<dim3(2, 72, 8), 256, 0, stream>>>(pcT, R4h, R1h,
          CK, Nn, Pp, Pp, Pp, Nn, sPC, sSim, sKey, 1, nullptr, nullptr, nullptr);

  // --- context = W_w @ x_ctx + W_b  (512x9216, K=256), fp32 to d_out ---
  gemm_wmma<DT_F32, DT_BF16, LAY_NN, EPI_BIAS, DT_F32, false>
      <<<dim3(4, 72, 8), 256, 0, stream>>>(W_w, R1h, d_out,
          CIN, Nn, CK, CK, Nn, Nn, 0, sKey, sX, 1, W_b, nullptr, nullptr);
}
